// GraphONE_30837865185503
// MI455X (gfx1250) — matrix operand order repaired and checked by reading, not compiled, not run on hardware
//
#include <hip/hip_runtime.h>

// ---------------- problem constants (match reference) ----------------
#define B_N   16384
#define K_N   2048
#define F_N   1024
#define H_N   1024
#define D_N   3
#define TOPK  8
#define LN_EPS 1e-5f

typedef __attribute__((ext_vector_type(16))) _Float16 v16h;
typedef __attribute__((ext_vector_type(8)))  _Float16 v8h;
typedef __attribute__((ext_vector_type(8)))  float    v8f;

// ---------------- optional CDNA5 async global->LDS path ----------------
#ifndef HAVE_ASYNC_LDS
#if defined(__has_builtin)
#if __has_builtin(__builtin_amdgcn_global_load_async_to_lds_b128)
#define HAVE_ASYNC_LDS 1
#endif
#endif
#endif
#ifndef HAVE_ASYNC_LDS
#define HAVE_ASYNC_LDS 0
#endif

#if HAVE_ASYNC_LDS
// Builtin signature (from hipcc diagnostic): arg0 = pointer to
// int __attribute__((vector_size(16))) in global (__device__, AS1) space,
// arg1 = same pointee in LDS (__shared__, AS3) space, then imm offset, imm cpol.
typedef int v4i_t __attribute__((vector_size(16)));
typedef v4i_t __attribute__((address_space(1)))* g4p;
typedef v4i_t __attribute__((address_space(3)))* l4p;
__device__ __forceinline__ void async_copy_b128(void* lds_dst, const void* gsrc) {
  // global_load_async_to_lds_b128: 16 bytes per lane, tracked by ASYNCcnt
  __builtin_amdgcn_global_load_async_to_lds_b128((g4p)gsrc, (l4p)lds_dst, 0, 0);
}
__device__ __forceinline__ void async_wait_all() {
#if defined(__has_builtin) && __has_builtin(__builtin_amdgcn_s_wait_asynccnt)
  __builtin_amdgcn_s_wait_asynccnt(0);
#else
  asm volatile("s_wait_asynccnt 0" ::: "memory");
#endif
}
#else
__device__ __forceinline__ void async_copy_b128(void* lds_dst, const void* gsrc) {
  *(v8h*)lds_dst = *(const v8h*)gsrc;   // fallback: VGPR round-trip
}
__device__ __forceinline__ void async_wait_all() {}
#endif

__device__ __forceinline__ v16h cat8(v8h lo, v8h hi) {
  return __builtin_shufflevector(lo, hi, 0,1,2,3,4,5,6,7,8,9,10,11,12,13,14,15);
}

__device__ __forceinline__ v8f wmma_f16(v16h a, v16h b, v8f c) {
  // D = A(16x32 f16) * B(32x16 f16) + C(16x16 f32)
  return __builtin_amdgcn_wmma_f32_16x16x32_f16(
      /*neg_a=*/false, a, /*neg_b=*/false, b,
      /*c_mod=*/(short)0, c, /*reuse_a=*/false, /*reuse_b=*/false);
}

// A-tile fetch from an LDS row-major [16][ldk] f16 buffer, matching the
// B operand K order: lanes 0-15 hold row m, K = kk+0..15;
// lanes 16-31 hold K = kk+16..31.
__device__ __forceinline__ v16h load_a_tile(const _Float16* As, int lane,
                                            int kk, int ldk) {
  int m  = lane & 15;
  int hi = (lane >> 4) * 16;
  const _Float16* r = As + m * ldk + kk;
  v8h lo8 = *(const v8h*)(r + hi);
  v8h hi8 = *(const v8h*)(r + hi + 8);
  return cat8(lo8, hi8);
}

// B-tile fetch from global: B[k][n] = W[n][k] (weight rows contiguous in K).
// B K order used here (contiguous, same as A, not the ISA table order): lanes 0-15 hold column n with K=kk+0..15,
// lanes 16-31 hold K=kk+16..31 (contiguous 32-byte load per lane).
__device__ __forceinline__ v16h load_b_tile(const _Float16* Bm, int lane,
                                            int nb, int kk, int ldk) {
  int n  = nb + (lane & 15);
  int ko = kk + ((lane >> 4) << 4);
  return *(const v16h*)(Bm + (size_t)n * ldk + ko);
}

// ---------------- prologue: row-normalize + f32->f16 ----------------
__global__ void k_rownorm_f16(const float* __restrict__ src,
                              _Float16* __restrict__ dst, int ncols) {
  __shared__ float red[8];
  const int row = blockIdx.x, tid = threadIdx.x;
  const int wave = tid >> 5, lane = tid & 31;
  const float* s = src + (size_t)row * ncols;
  float p = 0.f;
  for (int c = tid; c < ncols; c += blockDim.x) { float v = s[c]; p += v * v; }
#pragma unroll
  for (int off = 16; off > 0; off >>= 1) p += __shfl_xor(p, off, 32);
  if (lane == 0) red[wave] = p;
  __syncthreads();
  float tot = 0.f;
#pragma unroll
  for (int i = 0; i < 8; ++i) tot += red[i];
  float sc = rsqrtf(tot > 1e-30f ? tot : 1e-30f);   // 1 / ||row||
  _Float16* d = dst + (size_t)row * ncols;
  for (int c = tid; c < ncols; c += blockDim.x) d[c] = (_Float16)(s[c] * sc);
}

__global__ void k_convert_f16(const float* __restrict__ src,
                              _Float16* __restrict__ dst, long long n) {
  long long i = (long long)blockIdx.x * blockDim.x + threadIdx.x;
  long long stride = (long long)gridDim.x * blockDim.x;
  for (; i < n; i += stride) dst[i] = (_Float16)src[i];
}

// ---------------- sim GEMM + fused per-row top-8 in LDS ----------------
// One workgroup: 16 feature rows x all 2048 prototypes. sim strip (128KB)
// never touches HBM. 8 waves, 16 N-tiles each; K = F_N via 16x16x32 WMMA.
__global__ void k_sim_topk(const _Float16* __restrict__ fn16,
                           const _Float16* __restrict__ gn16,
                           int* __restrict__ closest) {
  extern __shared__ char smem[];
  _Float16* As   = (_Float16*)smem;                                 // 16x1024 f16 (32KB)
  float*    simb = (float*)(smem + (size_t)16 * F_N * sizeof(_Float16)); // 16x2048 f32 (128KB)

  const int tid = threadIdx.x, wave = tid >> 5, lane = tid & 31;
  const int mBase = blockIdx.x * 16;

  // stage normalized feature rows into LDS (async global->LDS, 16B/lane)
  for (int i = tid; i < 16 * (F_N / 8); i += blockDim.x) {
    int row = i / (F_N / 8);
    int c8  = (i % (F_N / 8)) * 8;
    async_copy_b128(As + row * F_N + c8,
                    fn16 + (size_t)(mBase + row) * F_N + c8);
  }
  async_wait_all();
  __syncthreads();

  for (int t = wave * 16; t < wave * 16 + 16; ++t) {
    const int nb = t * 16;
    v8f acc = {};
    for (int kk = 0; kk < F_N; kk += 32) {
      v16h a = load_a_tile(As, lane, kk, F_N);
      v16h b = load_b_tile(gn16, lane, nb, kk, F_N);
      acc = wmma_f16(a, b, acc);
    }
    // scatter the 16x16 C tile into the LDS sim strip
    const int col = nb + (lane & 15);
    const int rbase = (lane >> 4) * 8;
#pragma unroll
    for (int r = 0; r < 8; ++r) simb[(rbase + r) * K_N + col] = acc[r];
  }
  __syncthreads();

  // top-8 per row: wave w owns rows 2w, 2w+1; iterative argmax extraction
  for (int rr = 0; rr < 2; ++rr) {
    const int row = wave * 2 + rr;
    float* srow = simb + row * K_N;
    for (int it = 0; it < TOPK; ++it) {
      float bv = -1e30f; int bi = 0;
      for (int c = lane; c < K_N; c += 32) {
        float v = srow[c];
        if (v > bv) { bv = v; bi = c; }
      }
#pragma unroll
      for (int off = 16; off > 0; off >>= 1) {
        float ov = __shfl_xor(bv, off, 32);
        int   oi = __shfl_xor(bi, off, 32);
        if (ov > bv || (ov == bv && oi < bi)) { bv = ov; bi = oi; }
      }
      if (lane == 0) {
        closest[(size_t)(mBase + row) * TOPK + it] = bi;
        srow[bi] = -1e30f;            // knock out for next extraction
      }
      // wave-ordered LDS: subsequent loads in this wave see the store
    }
  }
}

// ---------------- gather + max over top-k prototype rows ----------------
__global__ void k_aggproto(const float* __restrict__ graphone,
                           const int* __restrict__ closest,
                           _Float16* __restrict__ agg16,
                           int* __restrict__ assign_out) {
  const int b = blockIdx.x, tid = threadIdx.x;
  int idx[TOPK];
#pragma unroll
  for (int j = 0; j < TOPK; ++j) idx[j] = closest[(size_t)b * TOPK + j];
  for (int c = tid; c < F_N; c += blockDim.x) {
    float m = -1e30f;
#pragma unroll
    for (int j = 0; j < TOPK; ++j) {
      float v = graphone[(size_t)idx[j] * F_N + c];
      m = v > m ? v : m;
    }
    agg16[(size_t)b * F_N + c] = (_Float16)m;
  }
  if (tid == 0) {
#pragma unroll
    for (int d = 0; d < D_N; ++d) assign_out[(size_t)d * B_N + b] = idx[0];
  }
}

// ---------------- SAGE GEMM (h = max(agg,x)@Wl^T + x@Wr^T) + fused LN+ReLU
// One workgroup: 16 rows x full H=1024. The 64KB A staging buffer in LDS is
// reused as the f32 h buffer for the LayerNorm row reduction after a barrier.
__global__ void k_gemm_h_ln(const _Float16* __restrict__ agg16,
                            const _Float16* __restrict__ x16,
                            const _Float16* __restrict__ wl,
                            const _Float16* __restrict__ wr,
                            const float* __restrict__ gamma,
                            const float* __restrict__ beta,
                            _Float16* __restrict__ h16) {
  extern __shared__ char smem[];
  _Float16* Aagg = (_Float16*)smem;          // 16x1024 f16 (32KB): max(agg_proto, x)
  _Float16* Ax   = Aagg + 16 * F_N;          // 16x1024 f16 (32KB): x
  float*    hbuf = (float*)smem;             // reused: 16x1024 f32 (64KB)

  const int tid = threadIdx.x, wave = tid >> 5, lane = tid & 31;
  const int mBase = blockIdx.x * 16;

  // stage both source arrays with async global->LDS copies
  for (int i = tid; i < 16 * (F_N / 8); i += blockDim.x) {
    int row = i / (F_N / 8);
    int c8  = (i % (F_N / 8)) * 8;
    size_t g = (size_t)(mBase + row) * F_N + c8;
    async_copy_b128(Ax   + row * F_N + c8, x16   + g);
    async_copy_b128(Aagg + row * F_N + c8, agg16 + g);
  }
  async_wait_all();
  __syncthreads();

  // fuse agg = max(agg_proto, x) in LDS
  for (int i = tid; i < 16 * (F_N / 8); i += blockDim.x) {
    int o = i * 8;
    v8h xa = *(const v8h*)(Ax + o);
    v8h pa = *(const v8h*)(Aagg + o);
    v8h mx;
#pragma unroll
    for (int e = 0; e < 8; ++e) mx[e] = (pa[e] > xa[e]) ? pa[e] : xa[e];
    *(v8h*)(Aagg + o) = mx;
  }
  __syncthreads();

  v8f acc[8];
  for (int t = 0; t < 8; ++t) {
    const int nb = (wave * 8 + t) * 16;
    v8f c = {};
    for (int kk = 0; kk < F_N; kk += 32) {
      if (kk + 32 < F_N)
        __builtin_prefetch(wl + (size_t)(nb + (lane & 15)) * F_N + kk + 32, 0, 1);
      v16h aA = load_a_tile(Aagg, lane, kk, F_N);
      v16h aX = load_a_tile(Ax,   lane, kk, F_N);
      v16h bL = load_b_tile(wl, lane, nb, kk, F_N);
      v16h bR = load_b_tile(wr, lane, nb, kk, F_N);
      c = wmma_f16(aA, bL, c);
      c = wmma_f16(aX, bR, c);
    }
    acc[t] = c;
  }
  __syncthreads();                            // done reading A; reuse LDS as hbuf

  for (int t = 0; t < 8; ++t) {
    const int nb = (wave * 8 + t) * 16;
    const int col = nb + (lane & 15);
    const int rbase = (lane >> 4) * 8;
#pragma unroll
    for (int r = 0; r < 8; ++r) hbuf[(rbase + r) * H_N + col] = acc[t][r];
  }
  __syncthreads();

  // LayerNorm + ReLU; wave w owns rows 2w, 2w+1 (32-lane row reduction)
  for (int rr = 0; rr < 2; ++rr) {
    const int row = wave * 2 + rr;
    const float* hr = hbuf + row * H_N;
    float s1 = 0.f, s2 = 0.f;
    for (int c = lane; c < H_N; c += 32) { float v = hr[c]; s1 += v; s2 += v * v; }
#pragma unroll
    for (int off = 16; off > 0; off >>= 1) {
      s1 += __shfl_xor(s1, off, 32);
      s2 += __shfl_xor(s2, off, 32);
    }
    const float mu  = s1 * (1.f / H_N);
    const float var = s2 * (1.f / H_N) - mu * mu;
    const float rs  = rsqrtf(var + LN_EPS);
    _Float16* dst = h16 + (size_t)(mBase + row) * H_N;
    for (int c = lane; c < H_N; c += 32) {
      float v = (hr[c] - mu) * rs * gamma[c] + beta[c];
      v = v > 0.f ? v : 0.f;
      dst[c] = (_Float16)v;
    }
  }
}

// ---------------- projection GEMM: x = h @ Wp^T + b_p ----------------
__global__ void k_gemm_proj(const _Float16* __restrict__ h16,
                            const _Float16* __restrict__ wp,
                            const float* __restrict__ bp,
                            float* __restrict__ xout,
                            _Float16* __restrict__ x16out) {
  extern __shared__ char smem[];
  _Float16* As = (_Float16*)smem;             // 16x1024 f16 (32KB)
  const int tid = threadIdx.x, wave = tid >> 5, lane = tid & 31;
  const int mBase = blockIdx.x * 16;

  for (int i = tid; i < 16 * (H_N / 8); i += blockDim.x) {
    int row = i / (H_N / 8);
    int c8  = (i % (H_N / 8)) * 8;
    async_copy_b128(As + row * H_N + c8,
                    h16 + (size_t)(mBase + row) * H_N + c8);
  }
  async_wait_all();
  __syncthreads();

  for (int t = 0; t < 8; ++t) {
    const int nb = (wave * 8 + t) * 16;
    v8f c = {};
    for (int kk = 0; kk < H_N; kk += 32) {
      if (kk + 32 < H_N)
        __builtin_prefetch(wp + (size_t)(nb + (lane & 15)) * H_N + kk + 32, 0, 1);
      v16h a = load_a_tile(As, lane, kk, H_N);
      v16h b = load_b_tile(wp, lane, nb, kk, H_N);
      c = wmma_f16(a, b, c);
    }
    const int col = nb + (lane & 15);
    const int rbase = (lane >> 4) * 8;
    const float bias = bp[col];
#pragma unroll
    for (int r = 0; r < 8; ++r) {
      float v = c[r] + bias;
      size_t o = (size_t)(mBase + rbase + r) * F_N + col;
      xout[o] = v;                 // f32 (d_out; final depth leaves the answer)
      x16out[o] = (_Float16)v;     // f16 feed for next depth
    }
  }
}

// ---------------- host orchestration ----------------
extern "C" void kernel_launch(void* const* d_in, const int* in_sizes, int n_in,
                              void* d_out, int out_size, void* d_ws, size_t ws_size,
                              hipStream_t stream) {
  (void)in_sizes; (void)n_in; (void)out_size; (void)ws_size;
  const float* features = (const float*)d_in[0];
  const float* graphone = (const float*)d_in[1];
  const float* W_l      = (const float*)d_in[2];
  const float* W_r      = (const float*)d_in[3];
  const float* gamma    = (const float*)d_in[4];
  const float* beta     = (const float*)d_in[5];
  const float* W_p      = (const float*)d_in[6];
  const float* b_p      = (const float*)d_in[7];

  char* ws = (char*)d_ws;
  size_t off = 0;
  auto walloc = [&](size_t bytes) -> char* {
    char* p = ws + off;
    off += (bytes + 255) & ~(size_t)255;
    return p;
  };
  _Float16* fn16  = (_Float16*)walloc((size_t)B_N * F_N * 2);
  _Float16* gn16  = (_Float16*)walloc((size_t)K_N * F_N * 2);
  _Float16* x16   = (_Float16*)walloc((size_t)B_N * F_N * 2);
  _Float16* agg16 = (_Float16*)walloc((size_t)B_N * F_N * 2);
  _Float16* h16   = (_Float16*)walloc((size_t)B_N * H_N * 2);
  _Float16* wl16  = (_Float16*)walloc((size_t)D_N * H_N * F_N * 2);
  _Float16* wr16  = (_Float16*)walloc((size_t)D_N * H_N * F_N * 2);
  _Float16* wp16  = (_Float16*)walloc((size_t)D_N * F_N * H_N * 2);
  int*      closest = (int*)walloc((size_t)B_N * TOPK * 4);

  // prologue conversions
  k_rownorm_f16<<<B_N, 256, 0, stream>>>(features, fn16, F_N);
  k_rownorm_f16<<<K_N, 256, 0, stream>>>(graphone, gn16, F_N);
  k_convert_f16<<<4096, 256, 0, stream>>>(features, x16, (long long)B_N * F_N);
  k_convert_f16<<<4096, 256, 0, stream>>>(W_l, wl16, (long long)D_N * H_N * F_N);
  k_convert_f16<<<4096, 256, 0, stream>>>(W_r, wr16, (long long)D_N * H_N * F_N);
  k_convert_f16<<<4096, 256, 0, stream>>>(W_p, wp16, (long long)D_N * F_N * H_N);

  // sim GEMM + fused top-k (sim strip stays in LDS: 32KB A + 128KB sim)
  size_t simLds = (size_t)16 * F_N * 2 + (size_t)16 * K_N * 4;
  k_sim_topk<<<B_N / 16, 256, simLds, stream>>>(fn16, gn16, closest);

  int* assign_out = (int*)((float*)d_out + (size_t)B_N * F_N);
  k_aggproto<<<B_N, 256, 0, stream>>>(graphone, closest, agg16, assign_out);

  for (int d = 0; d < D_N; ++d) {
    k_gemm_h_ln<<<B_N / 16, 256, (size_t)16 * F_N * 2 * 2, stream>>>(
        agg16, x16,
        wl16 + (size_t)d * H_N * F_N, wr16 + (size_t)d * H_N * F_N,
        gamma + (size_t)d * H_N, beta + (size_t)d * H_N, h16);
    k_gemm_proj<<<B_N / 16, 256, (size_t)16 * H_N * 2, stream>>>(
        h16, wp16 + (size_t)d * F_N * H_N, b_p + (size_t)d * F_N,
        (float*)d_out, x16);
  }
}
